// Attention_30150670418307
// MI455X (gfx1250) — matrix operand order repaired and checked by reading, not compiled, not run on hardware
//
#include <hip/hip_runtime.h>
#include <cstdint>

// ---------------------------------------------------------------------------
// MI455X (gfx1250) attention block:
//   x[8,1024,768] @ W_qkv[768,2304] -> per-head q,k,v -> flash attention ->
//   out @ W_proj[768,768] + b_proj
// Matmuls on v_wmma_f32_16x16x32_bf16 (f32 accumulate).
// Tile staging via GLOBAL_LOAD_ASYNC_TO_LDS_B128 + s_wait_asynccnt,
// double-buffered LDS (weights/activations pre-converted to bf16 once).
// ---------------------------------------------------------------------------

typedef __attribute__((ext_vector_type(16))) __bf16 v16bf;
typedef __attribute__((ext_vector_type(8)))  float  v8f;

#define NUM_HEADS 12
#define HEAD_DIM  64
#define DIM       768
#define SEQ       1024
#define BATCH     8
#define MROWS     (BATCH * SEQ)      // 8192
#define QK_SCALE  0.125f             // 64^-0.5

union FragU { v16bf v; uint4 q[2]; };

// ---- CDNA5 async copy: global -> LDS, 16B per lane, tracked by ASYNCcnt ----
__device__ inline void async_copy_b128(void* lds_ptr, const void* gptr) {
    uint32_t lds = (uint32_t)(uintptr_t)lds_ptr;   // LDS aperture: offset = low 32 bits
    uint64_t ga  = (uint64_t)(uintptr_t)gptr;
    asm volatile("global_load_async_to_lds_b128 %0, %1, off"
                 :: "v"(lds), "v"(ga) : "memory");
}
__device__ inline void wait_async_le4() { asm volatile("s_wait_asynccnt 0x4" ::: "memory"); }
__device__ inline void wait_async_le0() { asm volatile("s_wait_asynccnt 0x0" ::: "memory"); }

// A-matrix 16x32 bf16 fragment from LDS tile stored row-major [m][k].
// Lane layout (ISA 7.12.2): row = lane%16, kh = lane/16;
// elems 0..7 <- K = kh*8 + 0..7 ; elems 8..15 <- K = 16 + kh*8 + 0..7
__device__ inline v16bf load_frag_a(const __bf16* base, int stride, int lane) {
    const int row = lane & 15;
    const int kh  = (lane >> 4) & 1;
    const __bf16* r = base + row * stride + kh * 8;
    FragU u;
    __builtin_memcpy(&u.q[0], r, 16);
    __builtin_memcpy(&u.q[1], r + 16, 16);
    return u.v;
}

// B-matrix 32x16 bf16 fragment from LDS tile stored [n][k].
// Lane layout: col n = lane%16, kh = lane/16; elems 0..7 <- K = kh*8 + 0..7 ; elems 8..15 <- K = 16 + kh*8 + 0..7.
__device__ inline v16bf load_frag_b(const __bf16* base, int stride, int lane) {
    const int n  = lane & 15;
    const int kh = (lane >> 4) & 1;
    const __bf16* r = base + n * stride + kh * 8;
    FragU u;
    __builtin_memcpy(&u.q[0], r, 16);
    __builtin_memcpy(&u.q[1], r + 16, 16);
    return u.v;
}

// ---------------------------------------------------------------------------
// One-shot converters (hoist fp32->bf16 + weight transpose out of GEMM loops)
// ---------------------------------------------------------------------------
__global__ __launch_bounds__(256)
void cvt_kernel(const float* __restrict__ in, __bf16* __restrict__ out, int n4) {
    int i = blockIdx.x * 256 + threadIdx.x;
    if (i < n4) {
        float4 f = ((const float4*)in)[i];
        union { __bf16 h[4]; uint2 u; } pk;
        pk.h[0] = (__bf16)f.x; pk.h[1] = (__bf16)f.y;
        pk.h[2] = (__bf16)f.z; pk.h[3] = (__bf16)f.w;
        ((uint2*)out)[i] = pk.u;
    }
}

// W[K][N] fp32 -> Wt[N][K] bf16 (tiled through LDS: 32(K) x 64(N) tiles)
__global__ __launch_bounds__(256)
void tcvt_kernel(const float* __restrict__ W, __bf16* __restrict__ Wt, int K, int N) {
    __shared__ __align__(16) __bf16 T[64 * 40];
    const int tid = threadIdx.x;
    const int bk = blockIdx.x * 32, bn = blockIdx.y * 64;
#pragma unroll
    for (int i = 0; i < 2; ++i) {
        int idx = tid + 256 * i;                // 512 float4 = 32 rows x 16
        int kr  = idx >> 4;
        int c4  = (idx & 15) << 2;
        float4 f = *(const float4*)(W + (size_t)(bk + kr) * N + bn + c4);
        T[(c4 + 0) * 40 + kr] = (__bf16)f.x;
        T[(c4 + 1) * 40 + kr] = (__bf16)f.y;
        T[(c4 + 2) * 40 + kr] = (__bf16)f.z;
        T[(c4 + 3) * 40 + kr] = (__bf16)f.w;
    }
    __syncthreads();
    int r = tid >> 2, c8 = (tid & 3) << 3;      // 256 x 16B out
    __builtin_memcpy(Wt + (size_t)(bn + r) * K + bk + c8, &T[r * 40 + c8], 16);
}

// ---------------------------------------------------------------------------
// Async double-buffered GEMM: C = A[M,K](bf16) x Bt[N,K](bf16)
// MODE 0: epilogue scatters bf16 q (scaled), k, v^T.
// MODE 2: epilogue adds bias, writes fp32 d_out.
// Block 128x128, BK=32, 256 threads = 8 waves (2x4 accumulators each).
// ---------------------------------------------------------------------------
#define LDSK 40

__device__ inline void gemm_issue(const __bf16* A, const __bf16* Bt,
                                  __bf16* as, __bf16* bs,
                                  int bm, int bn, int K, int k0, int tid) {
#pragma unroll
    for (int i = 0; i < 2; ++i) {
        int idx = tid + 256 * i;                // 512 x 16B per tile
        int r   = idx >> 2;
        int c8  = (idx & 3) << 3;
        async_copy_b128(as + r * LDSK + c8, A  + (size_t)(bm + r) * K + k0 + c8);
        async_copy_b128(bs + r * LDSK + c8, Bt + (size_t)(bn + r) * K + k0 + c8);
    }
}

template<int MODE>
__global__ __launch_bounds__(256)
void gemm_kernel(const __bf16* __restrict__ A, const __bf16* __restrict__ Bt,
                 const float* __restrict__ bias, float* __restrict__ Cout,
                 __bf16* __restrict__ qb, __bf16* __restrict__ kb,
                 __bf16* __restrict__ vtb, int M, int K, int N) {
    const int tid  = threadIdx.x;
    const int lane = tid & 31;
    const int w    = tid >> 5;
    const int wm   = w & 3;
    const int wn   = w >> 2;
    const int bm   = blockIdx.x * 128;
    const int bn   = blockIdx.y * 128;

    __shared__ __align__(16) __bf16 As[2][128 * LDSK];   // [m][k]
    __shared__ __align__(16) __bf16 Bs[2][128 * LDSK];   // [n][k]

    const v8f vzero = {0.f, 0.f, 0.f, 0.f, 0.f, 0.f, 0.f, 0.f};
    v8f acc[2][4];
#pragma unroll
    for (int i = 0; i < 2; ++i)
#pragma unroll
        for (int j = 0; j < 4; ++j) acc[i][j] = vzero;

    const int nsteps = K >> 5;                 // 24
    gemm_issue(A, Bt, As[0], Bs[0], bm, bn, K, 0, tid);

    for (int s = 0; s < nsteps; ++s) {
        const int cur = s & 1;
        if (s + 1 < nsteps) {
            gemm_issue(A, Bt, As[1 - cur], Bs[1 - cur], bm, bn, K, (s + 1) << 5, tid);
            wait_async_le4();                  // current tile's 4 loads are done
        } else {
            wait_async_le0();
        }
        __syncthreads();

        v16bf bfrag[4];
#pragma unroll
        for (int j = 0; j < 4; ++j)
            bfrag[j] = load_frag_b(&Bs[cur][(wn * 64 + j * 16) * LDSK], LDSK, lane);
#pragma unroll
        for (int i = 0; i < 2; ++i) {
            v16bf afrag = load_frag_a(&As[cur][(wm * 32 + i * 16) * LDSK], LDSK, lane);
#pragma unroll
            for (int j = 0; j < 4; ++j)
                acc[i][j] = __builtin_amdgcn_wmma_f32_16x16x32_bf16(
                    false, afrag, false, bfrag[j], (short)0, acc[i][j], false, false);
        }
        __syncthreads();                       // reads done before buf reuse
    }

    const int n  = lane & 15;
    const int mh = lane >> 4;
#pragma unroll
    for (int i = 0; i < 2; ++i)
#pragma unroll
        for (int j = 0; j < 4; ++j)
#pragma unroll
            for (int v = 0; v < 8; ++v) {
                int row = bm + wm * 32 + i * 16 + mh * 8 + v;
                int col = bn + wn * 64 + j * 16 + n;
                float val = acc[i][j][v];
                if constexpr (MODE == 0) {
                    int b  = row >> 10, nn2 = row & 1023;
                    int sel = col / DIM, cc = col - sel * DIM;
                    int h  = cc >> 6,  d  = cc & 63;
                    size_t bh = (size_t)(b * NUM_HEADS + h);
                    if (sel == 0)      qb[(bh * SEQ + nn2) * HEAD_DIM + d] = (__bf16)(val * QK_SCALE);
                    else if (sel == 1) kb[(bh * SEQ + nn2) * HEAD_DIM + d] = (__bf16)val;
                    else               vtb[(bh * HEAD_DIM + d) * SEQ + nn2] = (__bf16)val;  // V^T
                } else {
                    Cout[(size_t)row * N + col] = val + bias[col];
                }
            }
}

// ---------------------------------------------------------------------------
// Flash attention: one WG per (b,h,128-row Q block); 8 waves, 16 rows each.
// 16 key blocks of 64, K/V staged with async copies, double-buffered.
// ---------------------------------------------------------------------------
#define LDSD 72

__device__ inline void flash_issue(const __bf16* kg, const __bf16* vg,
                                   __bf16* ks, __bf16* vs, int tid) {
#pragma unroll
    for (int i = 0; i < 2; ++i) {
        int idx = tid + 256 * i;               // 512 x 16B per tile
        int r   = idx >> 3;
        int c8  = (idx & 7) << 3;
        async_copy_b128(ks + r * LDSD + c8, kg + (size_t)r * HEAD_DIM + c8);
        async_copy_b128(vs + r * LDSD + c8, vg + (size_t)r * SEQ + c8);
    }
}

__global__ __launch_bounds__(256)
void flash_kernel(const __bf16* __restrict__ q, const __bf16* __restrict__ k,
                  const __bf16* __restrict__ vt, __bf16* __restrict__ o) {
    const int tid  = threadIdx.x;
    const int lane = tid & 31;
    const int w    = tid >> 5;
    const int bh   = blockIdx.y;
    const int qblk = blockIdx.x;
    const int b    = bh / NUM_HEADS;
    const int h    = bh - b * NUM_HEADS;

    __shared__ __align__(16) __bf16 Qs[128 * LDSD];        // [qrow][d]
    __shared__ __align__(16) __bf16 Ks[2][64 * LDSD];      // [key][d]
    __shared__ __align__(16) __bf16 Vs[2][64 * LDSD];      // [d][key] (V^T)
    __shared__ __align__(16) __bf16 Ps[8 * 16 * LDSD];     // per-wave P [16][64]

    // Q block (128 x 64), regular loads
    const __bf16* qg = q + ((size_t)bh * SEQ + qblk * 128) * HEAD_DIM;
#pragma unroll
    for (int i = 0; i < 4; ++i) {
        int idx = tid + 256 * i;
        int r   = idx >> 3;
        int c8  = (idx & 7) << 3;
        uint4 u;
        __builtin_memcpy(&u, qg + (size_t)r * HEAD_DIM + c8, 16);
        __builtin_memcpy(&Qs[r * LDSD + c8], &u, 16);
    }

    const __bf16* kbase = k  + (size_t)bh * SEQ * HEAD_DIM;
    const __bf16* vbase = vt + (size_t)bh * HEAD_DIM * SEQ;
    flash_issue(kbase, vbase, Ks[0], Vs[0], tid);

    const v8f vzero = {0.f, 0.f, 0.f, 0.f, 0.f, 0.f, 0.f, 0.f};
    v8f   oacc[4];
    float m8[8], l8[8];
#pragma unroll
    for (int t = 0; t < 4; ++t) oacc[t] = vzero;
#pragma unroll
    for (int v = 0; v < 8; ++v) { m8[v] = -3.0e38f; l8[v] = 0.f; }

    const int nn = lane & 15;
    const int mh = lane >> 4;
    __bf16* pw = &Ps[w * 16 * LDSD];

    for (int kb2 = 0; kb2 < 16; ++kb2) {
        const int cur = kb2 & 1;
        if (kb2 + 1 < 16) {
            flash_issue(kbase + (size_t)(kb2 + 1) * 64 * HEAD_DIM,
                        vbase + (kb2 + 1) * 64,
                        Ks[1 - cur], Vs[1 - cur], tid);
            wait_async_le4();
        } else {
            wait_async_le0();
        }
        __syncthreads();

        // S = Q K^T : 16 rows x 64 keys per wave (q pre-scaled)
        v8f sacc[4];
#pragma unroll
        for (int t = 0; t < 4; ++t) sacc[t] = vzero;
#pragma unroll
        for (int kk = 0; kk < 2; ++kk) {
            v16bf aq = load_frag_a(&Qs[(w * 16) * LDSD + kk * 32], LDSD, lane);
#pragma unroll
            for (int t = 0; t < 4; ++t) {
                v16bf bk = load_frag_b(&Ks[cur][(t * 16) * LDSD + kk * 32], LDSD, lane);
                sacc[t] = __builtin_amdgcn_wmma_f32_16x16x32_bf16(
                    false, aq, false, bk, (short)0, sacc[t], false, false);
            }
        }

        // online softmax (row stats live in the 16-lane half)
#pragma unroll
        for (int v = 0; v < 8; ++v) {
            float mx = fmaxf(fmaxf(sacc[0][v], sacc[1][v]),
                             fmaxf(sacc[2][v], sacc[3][v]));
#pragma unroll
            for (int off = 1; off < 16; off <<= 1)
                mx = fmaxf(mx, __shfl_xor(mx, off, 32));
            float mnew  = fmaxf(m8[v], mx);
            float alpha = __expf(m8[v] - mnew);
            float psum  = 0.f;
#pragma unroll
            for (int t = 0; t < 4; ++t) {
                float p = __expf(sacc[t][v] - mnew);
                pw[(mh * 8 + v) * LDSD + t * 16 + nn] = (__bf16)p;   // C-layout -> LDS
                psum += p;
            }
#pragma unroll
            for (int off = 1; off < 16; off <<= 1)
                psum += __shfl_xor(psum, off, 32);
            l8[v] = l8[v] * alpha + psum;
            m8[v] = mnew;
#pragma unroll
            for (int t = 0; t < 4; ++t) oacc[t][v] *= alpha;
        }

        // O += P V   (same-wave DS ops are in-order)
#pragma unroll
        for (int kk = 0; kk < 2; ++kk) {
            v16bf ap = load_frag_a(&pw[kk * 32], LDSD, lane);
#pragma unroll
            for (int t = 0; t < 4; ++t) {
                v16bf bv = load_frag_b(&Vs[cur][(t * 16) * LDSD + kk * 32], LDSD, lane);
                oacc[t] = __builtin_amdgcn_wmma_f32_16x16x32_bf16(
                    false, ap, false, bv, (short)0, oacc[t], false, false);
            }
        }
        __syncthreads();                       // all reads done before buf reuse
    }

    // write O / l as bf16 in [b][n][h*64+d] layout for the proj GEMM
#pragma unroll
    for (int t = 0; t < 4; ++t)
#pragma unroll
        for (int v = 0; v < 8; ++v) {
            int row = qblk * 128 + w * 16 + mh * 8 + v;
            float val = oacc[t][v] / l8[v];
            o[((size_t)b * SEQ + row) * DIM + h * HEAD_DIM + t * 16 + nn] = (__bf16)val;
        }
}

// ---------------------------------------------------------------------------
extern "C" void kernel_launch(void* const* d_in, const int* in_sizes, int n_in,
                              void* d_out, int out_size, void* d_ws, size_t ws_size,
                              hipStream_t stream) {
    const float* x     = (const float*)d_in[0];
    const float* Wqkv  = (const float*)d_in[3];
    const float* Wproj = (const float*)d_in[4];
    const float* bproj = (const float*)d_in[5];
    float* out = (float*)d_out;

    const size_t SZ = (size_t)MROWS * DIM * sizeof(__bf16);   // 12.58 MB
    char* ws = (char*)d_ws;
    __bf16* qb     = (__bf16*)(ws + 0 * SZ);
    __bf16* kb     = (__bf16*)(ws + 1 * SZ);
    __bf16* vtb    = (__bf16*)(ws + 2 * SZ);
    __bf16* attnb  = (__bf16*)(ws + 3 * SZ);
    __bf16* xb     = (__bf16*)(ws + 4 * SZ);
    __bf16* wqkvt  = (__bf16*)(ws + 5 * SZ);                  // [2304][768]
    __bf16* wprojt = (__bf16*)(ws + 5 * SZ + (size_t)3 * DIM * DIM * sizeof(__bf16));

    // 0) one-shot conversions: x -> bf16; W -> bf16 transposed [n][k]
    cvt_kernel<<<(MROWS * DIM / 4 + 255) / 256, 256, 0, stream>>>(x, xb, MROWS * DIM / 4);
    tcvt_kernel<<<dim3(DIM / 32, 3 * DIM / 64), 256, 0, stream>>>(Wqkv, wqkvt, DIM, 3 * DIM);
    tcvt_kernel<<<dim3(DIM / 32, DIM / 64), 256, 0, stream>>>(Wproj, wprojt, DIM, DIM);

    // 1) QKV projection -> q (scaled), k, v^T (bf16, per-head layouts)
    gemm_kernel<0><<<dim3(MROWS / 128, 3 * DIM / 128), 256, 0, stream>>>(
        xb, wqkvt, nullptr, nullptr, qb, kb, vtb, MROWS, DIM, 3 * DIM);

    // 2) flash attention
    flash_kernel<<<dim3(SEQ / 128, BATCH * NUM_HEADS), 256, 0, stream>>>(
        qb, kb, vtb, attnb);

    // 3) output projection + bias -> fp32 out
    gemm_kernel<2><<<dim3(MROWS / 128, DIM / 128), 256, 0, stream>>>(
        attnb, wprojt, bproj, out, nullptr, nullptr, nullptr, MROWS, DIM, DIM);
}
